// HardTripletLoss_75969381531801
// MI455X (gfx1250) — matrix operand order repaired and checked by reading, not compiled, not run on hardware
//
#include <hip/hip_runtime.h>
#include <cstdint>
#include <cstddef>

// ---------------- problem constants (reference: N=8192, D=256) --------------
#define NROWS 8192
#define DDIM  256
#define CSPLIT 8                 // column slices across gridDim.y
#define SLICE (NROWS / CSPLIT)   // 1024 columns per slice
#define MBLK  64                 // anchor rows per block (4 waves x 16 rows)

typedef __attribute__((ext_vector_type(8)))  float   v8f;
typedef __attribute__((ext_vector_type(8)))  __bf16  v8bf;
typedef __attribute__((ext_vector_type(16))) __bf16  v16bf;

// ------------------------------ helpers -------------------------------------
__device__ __forceinline__ uint32_t f32_to_bf16_rne(float x) {
    uint32_t u = __float_as_uint(x);
    uint32_t r = u + 0x7FFFu + ((u >> 16) & 1u);   // round-to-nearest-even
    return r >> 16;
}

// Build a v16bf fragment from two contiguous 8x bf16 (16-byte) runs.
__device__ __forceinline__ v16bf make_frag(const uint16_t* p0, const uint16_t* p1) {
    v8bf a = *(const v8bf*)p0;
    v8bf b = *(const v8bf*)p1;
    return __builtin_shufflevector(a, b, 0,1,2,3,4,5,6,7,8,9,10,11,12,13,14,15);
}

__device__ __forceinline__ v8f wmma_bf16(v16bf a, v16bf b, v8f c) {
    // D = A(16x32 bf16) x B(32x16 bf16) + C(16x16 f32)
    return __builtin_amdgcn_wmma_f32_16x16x32_bf16(
        /*neg_a=*/false, a, /*neg_b=*/false, b,
        /*c_mod=*/(short)0, c, /*reuse_a=*/false, /*reuse_b=*/false);
}

// --------------------- kernel 0: zero the scalar output ---------------------
__global__ void k_zero(float* out) { out[0] = 0.0f; }

// ------------- kernel 1: fp32 -> bf16 hi/lo split + row norms ---------------
// One wave per row; each lane handles 8 contiguous elements.
__global__ void k_split_norm(const float* __restrict__ src,
                             uint16_t* __restrict__ hi,
                             uint16_t* __restrict__ lo,
                             float* __restrict__ norm) {
    const int row  = blockIdx.x * 8 + (threadIdx.x >> 5);
    const int lane = threadIdx.x & 31;
    const float* p = src + (size_t)row * DDIM + lane * 8;

    float4 x0 = *(const float4*)p;
    float4 x1 = *(const float4*)(p + 4);
    float xs[8] = {x0.x, x0.y, x0.z, x0.w, x1.x, x1.y, x1.z, x1.w};

    uint32_t hb[8], lb[8];
    float s = 0.0f;
#pragma unroll
    for (int i = 0; i < 8; ++i) {
        float x = xs[i];
        uint32_t h = f32_to_bf16_rne(x);
        float hf = __uint_as_float(h << 16);
        float lv = x - hf;
        hb[i] = h;
        lb[i] = f32_to_bf16_rne(lv);
        s = fmaf(x, x, s);
    }
    uint4 ph = make_uint4(hb[0] | (hb[1] << 16), hb[2] | (hb[3] << 16),
                          hb[4] | (hb[5] << 16), hb[6] | (hb[7] << 16));
    uint4 pl = make_uint4(lb[0] | (lb[1] << 16), lb[2] | (lb[3] << 16),
                          lb[4] | (lb[5] << 16), lb[6] | (lb[7] << 16));
    *(uint4*)(hi + (size_t)row * DDIM + lane * 8) = ph;
    *(uint4*)(lo + (size_t)row * DDIM + lane * 8) = pl;

#pragma unroll
    for (int off = 16; off >= 1; off >>= 1) s += __shfl_xor(s, off, 32);
    if (lane == 0) norm[row] = s;
}

// ------ kernel 2: fused dual-GEMM (bf16x3) + masked row max/min epilogue ----
// grid = (NROWS/MBLK, CSPLIT), block = 128 (4 waves). Wave w owns rows
// [blockIdx.x*64 + w*16, +16). Column slice = blockIdx.y*1024 .. +1024.
__global__ __launch_bounds__(128, 1)
void k_triplet_main(const uint16_t* __restrict__ Ahi, const uint16_t* __restrict__ Alo,
                    const uint16_t* __restrict__ Phi, const uint16_t* __restrict__ Plo,
                    const uint16_t* __restrict__ Nhi, const uint16_t* __restrict__ Nlo,
                    const float* __restrict__ anorm, const float* __restrict__ pnorm,
                    const float* __restrict__ nnorm, const int* __restrict__ tgt,
                    float* __restrict__ pmax_part, float* __restrict__ nmin_part) {
    const int lane = threadIdx.x & 31;
    const int wave = threadIdx.x >> 5;
    const int rowBase   = blockIdx.x * MBLK + wave * 16;
    const int sliceBase = blockIdx.y * SLICE;
    const int nl   = lane & 15;   // N / M position within 16
    const int half = lane >> 4;   // 0 or 1

    // ---- preload resident A fragments (16 rows x 256 K, hi + lo) ----
    // A 16x32 bf16 k order used here (contiguous, B matches): lane holds row M=nl; lanes<16 K=[0,16),
    // lanes>=16 K=[16,32)  -> two contiguous 16B runs.
    const int cA = half * 16;
    const uint16_t* arow_h = Ahi + (size_t)(rowBase + nl) * DDIM;
    const uint16_t* arow_l = Alo + (size_t)(rowBase + nl) * DDIM;
    v16bf Ah[8], Al[8];
#pragma unroll
    for (int kc = 0; kc < 8; ++kc) {
        const int kb = kc * 32 + cA;
        Ah[kc] = make_frag(arow_h + kb, arow_h + kb + 8);
        Al[kc] = make_frag(arow_l + kb, arow_l + kb + 8);
    }

    // per-lane row constants: lane's 8 C rows are m = rowBase + r + half*8
    float anr[8]; int tgm[8]; float pmax[8], nmin[8];
#pragma unroll
    for (int r = 0; r < 8; ++r) {
        const int m = rowBase + r + half * 8;
        anr[r] = anorm[m];
        tgm[r] = tgt[m];
        pmax[r] = 0.0f;              // dist*mask max: 0 is the identity
        nmin[r] = 3.4e38f;
    }

    // B 32x16 bf16 k order used here (same as A): lane holds column N=nl, 16 consecutive K starting
    // at half*16 -> one contiguous 32B run per fragment.
    const int c2 = half * 16;

    for (int t = 0; t < SLICE / 16; ++t) {
        const int nB = sliceBase + t * 16 + nl;
        const uint16_t* ph = Phi + (size_t)nB * DDIM + c2;
        const uint16_t* pl = Plo + (size_t)nB * DDIM + c2;
        const uint16_t* nh = Nhi + (size_t)nB * DDIM + c2;
        const uint16_t* nn_ = Nlo + (size_t)nB * DDIM + c2;

        v8f cP = {}; v8f cN = {};
#pragma unroll
        for (int kc = 0; kc < 8; ++kc) {
            const int kb = kc * 32;
            v16bf Bph = make_frag(ph + kb, ph + kb + 8);
            v16bf Bpl = make_frag(pl + kb, pl + kb + 8);
            v16bf Bnh = make_frag(nh + kb, nh + kb + 8);
            v16bf Bnl = make_frag(nn_ + kb, nn_ + kb + 8);
            // bf16x3 split: hi*hi + lo*hi + hi*lo  (~fp32 accuracy)
            cP = wmma_bf16(Ah[kc], Bph, cP);
            cP = wmma_bf16(Al[kc], Bph, cP);
            cP = wmma_bf16(Ah[kc], Bpl, cP);
            cN = wmma_bf16(Ah[kc], Bnh, cN);
            cN = wmma_bf16(Al[kc], Bnh, cN);
            cN = wmma_bf16(Ah[kc], Bnl, cN);
        }

        // epilogue: d = sqrt(||a||^2 + ||b||^2 - 2 a.b); masked max / min
        const float pn = pnorm[nB];
        const float nnv = nnorm[nB];
        const int   tn  = tgt[nB];
#pragma unroll
        for (int r = 0; r < 8; ++r) {
            const bool same = (tgm[r] == tn);
            float dp = sqrtf(fmaxf(anr[r] + pn - 2.0f * cP[r], 1e-12f));
            if (same) pmax[r] = fmaxf(pmax[r], dp);
            float dn = sqrtf(fmaxf(anr[r] + nnv - 2.0f * cN[r], 1e-12f));
            dn += same ? 9999.9f : 0.0f;
            nmin[r] = fminf(nmin[r], dn);
        }
    }

    // reduce across the 16 lanes sharing each row (groups = lane halves)
#pragma unroll
    for (int r = 0; r < 8; ++r) {
        float pm = pmax[r], nm = nmin[r];
#pragma unroll
        for (int off = 1; off < 16; off <<= 1) {
            pm = fmaxf(pm, __shfl_xor(pm, off, 16));
            nm = fminf(nm, __shfl_xor(nm, off, 16));
        }
        if (nl == 0) {
            const int m = rowBase + r + half * 8;
            pmax_part[(size_t)m * CSPLIT + blockIdx.y] = pm;
            nmin_part[(size_t)m * CSPLIT + blockIdx.y] = nm;
        }
    }
}

// --------- kernel 3: combine slice partials, relu-margin, mean --------------
__global__ void k_finalize(const float* __restrict__ pmax_part,
                           const float* __restrict__ nmin_part,
                           float* __restrict__ out) {
    const int m = blockIdx.x * 256 + threadIdx.x;
    float pm = 0.0f, nm = 3.4e38f;
#pragma unroll
    for (int s = 0; s < CSPLIT; ++s) {
        pm = fmaxf(pm, pmax_part[(size_t)m * CSPLIT + s]);
        nm = fminf(nm, nmin_part[(size_t)m * CSPLIT + s]);
    }
    float loss = fmaxf(pm - nm + 0.2f, 0.0f);

    __shared__ float sd[256];
    sd[threadIdx.x] = loss;
    __syncthreads();
#pragma unroll
    for (int s = 128; s >= 1; s >>= 1) {
        if (threadIdx.x < s) sd[threadIdx.x] += sd[threadIdx.x + s];
        __syncthreads();
    }
    if (threadIdx.x == 0) atomicAdd(out, sd[0] * (1.0f / (float)NROWS));
}

// ------------------------------- launcher -----------------------------------
extern "C" void kernel_launch(void* const* d_in, const int* in_sizes, int n_in,
                              void* d_out, int out_size, void* d_ws, size_t ws_size,
                              hipStream_t stream) {
    const float* anchor   = (const float*)d_in[0];
    const float* positive = (const float*)d_in[1];
    const float* negative = (const float*)d_in[2];
    const int*   targets  = (const int*)d_in[3];
    float* out = (float*)d_out;

    // workspace layout
    char* w = (char*)d_ws;
    const size_t MATB = (size_t)NROWS * DDIM * sizeof(uint16_t); // 4 MiB
    uint16_t* a_hi = (uint16_t*)(w + 0 * MATB);
    uint16_t* a_lo = (uint16_t*)(w + 1 * MATB);
    uint16_t* p_hi = (uint16_t*)(w + 2 * MATB);
    uint16_t* p_lo = (uint16_t*)(w + 3 * MATB);
    uint16_t* n_hi = (uint16_t*)(w + 4 * MATB);
    uint16_t* n_lo = (uint16_t*)(w + 5 * MATB);
    char* w2 = w + 6 * MATB;
    float* anorm = (float*)(w2);
    float* pnorm = (float*)(w2 + 1 * NROWS * sizeof(float));
    float* nnorm = (float*)(w2 + 2 * NROWS * sizeof(float));
    float* pmax_part = (float*)(w2 + 3 * NROWS * sizeof(float));
    float* nmin_part = pmax_part + (size_t)NROWS * CSPLIT;

    k_zero<<<1, 1, 0, stream>>>(out);

    k_split_norm<<<NROWS / 8, 256, 0, stream>>>(anchor,   a_hi, a_lo, anorm);
    k_split_norm<<<NROWS / 8, 256, 0, stream>>>(positive, p_hi, p_lo, pnorm);
    k_split_norm<<<NROWS / 8, 256, 0, stream>>>(negative, n_hi, n_lo, nnorm);

    dim3 grid(NROWS / MBLK, CSPLIT);   // 128 x 8
    k_triplet_main<<<grid, 128, 0, stream>>>(a_hi, a_lo, p_hi, p_lo, n_hi, n_lo,
                                             anorm, pnorm, nnorm, targets,
                                             pmax_part, nmin_part);

    k_finalize<<<NROWS / 256, 256, 0, stream>>>(pmax_part, nmin_part, out);
}
